// DualGCN_20590073217487
// MI455X (gfx1250) — matrix operand order repaired and checked by reading, not compiled, not run on hardware
//
#include <hip/hip_runtime.h>

// ---------------------------------------------------------------------------
// Dual 2-layer GCN for MI455X (gfx1250, wave32).
// GEMMs:   v_wmma_f32_16x16x32_f16, B staged in LDS (padded vs bank conflicts),
//          one-deep software pipeline so wmma overlaps the next fragment load,
//          fused self-loop + bias epilogue.
// Aggreg.: wave-per-edge gather + global_atomic_add_f32 scatter. With 192MB L2
//          the 25MB feature matrices stay resident; scatter is L2-bound.
// ---------------------------------------------------------------------------

#define GCN_N   50000
#define GCN_E   600000
#define GCN_IN1 256
#define GCN_IN2 128
#define GCN_HID 128
#define GCN_OUT 64

typedef __attribute__((ext_vector_type(16))) _Float16 v16h;
typedef __attribute__((ext_vector_type(8)))  _Float16 v8h;
typedef __attribute__((ext_vector_type(4)))  _Float16 v4h;
typedef __attribute__((ext_vector_type(8)))  float    v8f;
typedef __attribute__((ext_vector_type(4)))  float    v4f;

// ---------------------------------------------------------------------------
// WMMA GEMM + fused epilogue:
//   H[M x NOUT]     = A[M x K] @ Bt^T           (raw h, scatter source)
//   Oinit[M x NOUT] = H * dinv[row]^2 + bias[col]  (self-loop + bias init)
// A is f16 row-major; Bt is f16 [NOUT x K] (transposed weights).
// One wave per 16-row strip, all NOUT/16 column tiles accumulated in regs.
// Bt staged into LDS once per block; rows padded +8 halves so the 16 lanes of
// a half-wave land on disjoint bank quads (stride K+8 halves -> 4 banks/row).
// ---------------------------------------------------------------------------
template <int K, int NOUT>
__global__ __launch_bounds__(128)
void gcn_gemm_wmma(const _Float16* __restrict__ A,
                   const _Float16* __restrict__ Bt,
                   const float* __restrict__ dinv,
                   const float* __restrict__ bias,
                   float* __restrict__ H,
                   float* __restrict__ Oinit,
                   int M)
{
    constexpr int NT = NOUT / 16;
    constexpr int KP = K + 8;                       // padded LDS row stride
    alignas(16) __shared__ _Float16 blds[NOUT * KP];

    // cooperative stage of Bt into (padded) LDS
    constexpr int CHUNKS = (NOUT * K) / 8;          // 8-half (16B) chunks
    for (int c = threadIdx.x; c < CHUNKS; c += blockDim.x) {
        int row = c / (K / 8);
        int col = (c % (K / 8)) * 8;
        *(v8h*)(&blds[row * KP + col]) = *(const v8h*)(Bt + (size_t)row * K + col);
    }
    __syncthreads();

    const int wave = blockIdx.x * (blockDim.x >> 5) + (threadIdx.x >> 5);
    const int lane = threadIdx.x & 31;
    if (wave * 16 >= M) return;                     // wave-uniform: EXEC all-1s
    const int r  = lane & 15;
    const int hi = lane >> 4;

    v8f c[NT] = {};

    const _Float16* arow  = A + (size_t)(wave * 16 + r) * K + hi * 8;
    const _Float16* bbase = &blds[(size_t)r * KP + hi * 8];

    auto loadA = [&](int kk) -> v16h {
        v8h lo = *(const v8h*)(arow + kk * 32);
        v8h hh = *(const v8h*)(arow + kk * 32 + 16);
        return __builtin_shufflevector(lo, hh,
                                       0,1,2,3,4,5,6,7,8,9,10,11,12,13,14,15);
    };
    auto loadB = [&](int kk, int n) -> v16h {
        const _Float16* p = bbase + (size_t)n * 16 * KP + kk * 32;
        v8h lo = *(const v8h*)(p);
        v8h hh = *(const v8h*)(p + 16);
        return __builtin_shufflevector(lo, hh,
                                       0,1,2,3,4,5,6,7,8,9,10,11,12,13,14,15);
    };

    // one-deep software pipeline: fetch fragment (kk,n)+1 before wmma (kk,n)
    v16h acur = loadA(0);
    v16h bcur = loadB(0, 0);
#pragma unroll
    for (int kk = 0; kk < K / 32; ++kk) {
#pragma unroll
        for (int n = 0; n < NT; ++n) {
            v16h bnext = bcur;
            v16h anext = acur;
            if (!(kk == K / 32 - 1 && n == NT - 1)) {
                bnext = loadB((n + 1 == NT) ? kk + 1 : kk,
                              (n + 1 == NT) ? 0 : n + 1);
                if (n + 1 == NT) anext = loadA(kk + 1);
            }
            c[n] = __builtin_amdgcn_wmma_f32_16x16x32_f16(
                       false, acur, false, bcur, (short)0, c[n], false, false);
            bcur = bnext;
            if (n + 1 == NT) acur = anext;
        }
    }

    // fused epilogue: C/D layout VGPR g -> row hi*8+g, column lane&15
    const int rbase = wave * 16 + hi * 8;
    float dv[8];
#pragma unroll
    for (int g = 0; g < 8; ++g) {
        float d = dinv[rbase + g];
        dv[g] = d * d;
    }
#pragma unroll
    for (int n = 0; n < NT; ++n) {
        const int col = n * 16 + r;
        const float bs = bias[col];
        size_t base = (size_t)rbase * NOUT + col;
#pragma unroll
        for (int g = 0; g < 8; ++g) {
            float v = c[n][g];
            H[base + (size_t)g * NOUT]     = v;
            Oinit[base + (size_t)g * NOUT] = v * dv[g] + bs;
        }
    }
}

// ------------------------- helper kernels ---------------------------------

__global__ void zero_f32(float* __restrict__ p, int n)
{
    int i = blockIdx.x * blockDim.x + threadIdx.x;
    if (i < n) p[i] = 0.0f;
}

__global__ void deg_accum(const int* __restrict__ dst, float* __restrict__ deg, int e)
{
    int i = blockIdx.x * blockDim.x + threadIdx.x;
    if (i < e) atomicAdd(&deg[dst[i]], 1.0f);
}

__global__ void deg_to_dinv(float* __restrict__ deg, int n)
{
    int i = blockIdx.x * blockDim.x + threadIdx.x;
    if (i < n) deg[i] = rsqrtf(deg[i] + 1.0f);     // includes self-loop
}

// vectorized f32 -> f16 (4 elements / thread)
__global__ void f32_to_f16_v4(const float* __restrict__ x, _Float16* __restrict__ y,
                              size_t n4)
{
    size_t i = (size_t)blockIdx.x * blockDim.x + threadIdx.x;
    if (i < n4) {
        v4f v = *(const v4f*)(x + i * 4);
        *(v4h*)(y + i * 4) = __builtin_convertvector(v, v4h);
    }
}

// vectorized relu + f32 -> f16
__global__ void relu_to_f16_v4(const float* __restrict__ x, _Float16* __restrict__ y,
                               size_t n4)
{
    size_t i = (size_t)blockIdx.x * blockDim.x + threadIdx.x;
    if (i < n4) {
        v4f v = *(const v4f*)(x + i * 4);
#pragma unroll
        for (int j = 0; j < 4; ++j) v[j] = fmaxf(v[j], 0.0f);
        *(v4h*)(y + i * 4) = __builtin_convertvector(v, v4h);
    }
}

// Wt[n][k] = (f16) W[k][n] so the WMMA B operand loads contiguously.
__global__ void wt_convert(const float* __restrict__ W, _Float16* __restrict__ Wt,
                           int k_dim, int n_dim)
{
    int i = blockIdx.x * blockDim.x + threadIdx.x;
    if (i >= k_dim * n_dim) return;
    int k = i / n_dim, n = i % n_dim;
    Wt[(size_t)n * k_dim + k] = (_Float16)W[i];
}

// One wave per edge: coef = dinv[src]*dinv[dst]; out[dst] += h[src]*coef.
template <int F, int VEC>   // VEC = F/32 floats per lane
__global__ void edge_scatter(const float* __restrict__ h,
                             const int* __restrict__ src,
                             const int* __restrict__ dst,
                             const float* __restrict__ dinv,
                             float* __restrict__ out, int e)
{
    int wid  = (int)(((size_t)blockIdx.x * blockDim.x + threadIdx.x) >> 5);
    int lane = threadIdx.x & 31;
    if (wid >= e) return;                           // wave-uniform exit
    int s = src[wid];
    int d = dst[wid];
    float coef = dinv[s] * dinv[d];
    const float* hp = h   + (size_t)s * F + lane * VEC;
    float*       op = out + (size_t)d * F + lane * VEC;
#pragma unroll
    for (int j = 0; j < VEC; ++j)
        atomicAdd(&op[j], hp[j] * coef);
}

// ------------------------- workspace layout (bytes) -----------------------
// Fused epilogue => x_f16, h and o1 live simultaneously: three 25.6MB regions.
//   R1: x_f16 -> act_f16        R2: h1 -> h2        R3: o1
static const size_t OFF_DINV = 0;                       // N*4      = 200000
static const size_t OFF_W0T  = 204800;                  // <= 65536
static const size_t OFF_W1T  = 270336;                  // <= 16384
static const size_t OFF_R1   = 327680;                  // 25,600,000
static const size_t OFF_R2   = OFF_R1 + 25600000;       // 25,600,000
static const size_t OFF_R3   = OFF_R2 + 25600000;       // 25,600,000
// total ~77.2 MB

template <int IN>
static void run_graph(const float* x, const int* edge,
                      const float* W0, const float* b0,
                      const float* W1, const float* b1,
                      float* out, char* ws, hipStream_t stream)
{
    const int N = GCN_N, E = GCN_E;
    const int* src = edge;
    const int* dst = edge + E;

    float*    dinv = (float*)(ws + OFF_DINV);
    _Float16* w0t  = (_Float16*)(ws + OFF_W0T);
    _Float16* w1t  = (_Float16*)(ws + OFF_W1T);
    _Float16* xh   = (_Float16*)(ws + OFF_R1);
    _Float16* act  = (_Float16*)(ws + OFF_R1);      // xh dead after gemm1
    float*    h1   = (float*)(ws + OFF_R2);
    float*    h2   = (float*)(ws + OFF_R2);         // h1 dead after scatter1
    float*    o1   = (float*)(ws + OFF_R3);

    // --- normalization coefficients (needed by the gemm epilogue) ---
    zero_f32<<<(N + 255) / 256, 256, 0, stream>>>(dinv, N);
    deg_accum<<<(E + 255) / 256, 256, 0, stream>>>(dst, dinv, E);
    deg_to_dinv<<<(N + 255) / 256, 256, 0, stream>>>(dinv, N);

    // --- precision conversion for WMMA ---
    {
        size_t n4 = (size_t)N * IN / 4;
        f32_to_f16_v4<<<(unsigned)((n4 + 255) / 256), 256, 0, stream>>>(x, xh, n4);
    }
    wt_convert<<<(IN * GCN_HID + 255) / 256, 256, 0, stream>>>(W0, w0t, IN, GCN_HID);
    wt_convert<<<(GCN_HID * GCN_OUT + 255) / 256, 256, 0, stream>>>(W1, w1t, GCN_HID, GCN_OUT);

    const int waves = N / 16;                       // 3125, exact
    const int gblk  = (waves + 3) / 4;              // 4 waves / 128-thread block

    // --- layer 1: h1 = x @ W0 ; o1 = h1*dinv^2 + b0 (fused) ---
    gcn_gemm_wmma<IN, GCN_HID><<<gblk, 128, 0, stream>>>(
        xh, w0t, dinv, b0, h1, o1, N);
    // o1[dst] += h1[src]*coef
    edge_scatter<GCN_HID, GCN_HID / 32>
        <<<(unsigned)(((size_t)E * 32 + 255) / 256), 256, 0, stream>>>(
            h1, src, dst, dinv, o1, E);
    // act = relu(o1) as f16
    {
        size_t n4 = (size_t)N * GCN_HID / 4;
        relu_to_f16_v4<<<(unsigned)((n4 + 255) / 256), 256, 0, stream>>>(o1, act, n4);
    }

    // --- layer 2: h2 = act @ W1 ; out = h2*dinv^2 + b1 (fused) ---
    gcn_gemm_wmma<GCN_HID, GCN_OUT><<<gblk, 128, 0, stream>>>(
        act, w1t, dinv, b1, h2, out, N);
    edge_scatter<GCN_OUT, GCN_OUT / 32>
        <<<(unsigned)(((size_t)E * 32 + 255) / 256), 256, 0, stream>>>(
            h2, src, dst, dinv, out, E);
}

extern "C" void kernel_launch(void* const* d_in, const int* in_sizes, int n_in,
                              void* d_out, int out_size, void* d_ws, size_t ws_size,
                              hipStream_t stream)
{
    (void)in_sizes; (void)n_in; (void)out_size; (void)ws_size;

    const float* x1   = (const float*)d_in[0];
    const int*   ei1  = (const int*)  d_in[1];
    const float* x2   = (const float*)d_in[2];
    const int*   ei2  = (const int*)  d_in[3];
    const float* W1_0 = (const float*)d_in[4];
    const float* b1_0 = (const float*)d_in[5];
    const float* W1_1 = (const float*)d_in[6];
    const float* b1_1 = (const float*)d_in[7];
    const float* W2_0 = (const float*)d_in[8];
    const float* b2_0 = (const float*)d_in[9];
    const float* W2_1 = (const float*)d_in[10];
    const float* b2_1 = (const float*)d_in[11];

    float* out  = (float*)d_out;
    char*  ws   = (char*)d_ws;
    float* out1 = out;
    float* out2 = out + (size_t)GCN_N * GCN_OUT;

    run_graph<GCN_IN1>(x1, ei1, W1_0, b1_0, W1_1, b1_1, out1, ws, stream);
    run_graph<GCN_IN2>(x2, ei2, W2_0, b2_0, W2_1, b2_1, out2, ws, stream);
}
